// TestModel_11879879541231
// MI455X (gfx1250) — matrix operand order repaired and checked by reading, not compiled, not run on hardware
//
#include <hip/hip_runtime.h>
#include <hip/hip_bf16.h>

typedef __attribute__((ext_vector_type(16))) __bf16 v16bf;
typedef __attribute__((ext_vector_type(8)))  __bf16 v8bf;
typedef __attribute__((ext_vector_type(8)))  float  v8f;

constexpr int Bsz = 256;
constexpr int Tn  = 1024;
constexpr int Vn  = 6;
constexpr int Dn  = 64;
constexpr int Un  = 64;
constexpr int G4  = 256;   // 4*U gate columns

static __device__ __forceinline__ __bf16 f2bf(float f) {
    // round-to-nearest-even float32 -> bf16 (bit manipulation; portable)
    unsigned u = __builtin_bit_cast(unsigned, f);
    unsigned r = (u + 0x7FFFu + ((u >> 16) & 1u)) >> 16;
    return __builtin_bit_cast(__bf16, (unsigned short)r);
}
static __device__ __forceinline__ float bf2f(__bf16 b) {
    unsigned u = (unsigned)__builtin_bit_cast(unsigned short, b) << 16;
    return __builtin_bit_cast(float, u);
}

// branch-free fast activations on raw v_exp_f32 / v_rcp_f32 (TRANS ops co-execute with WMMA)
static __device__ __forceinline__ float fast_sigmoid(float x) {
    float e = __builtin_amdgcn_exp2f(x * -1.4426950408889634f);
    return __builtin_amdgcn_rcpf(1.0f + e);
}
static __device__ __forceinline__ float fast_tanh(float x) {
    float e = __builtin_amdgcn_exp2f(x * -2.8853900817779268f);
    float t = __builtin_amdgcn_rcpf(1.0f + e);
    return __builtin_fmaf(2.0f, t, -1.0f);
}

// -------- Precompute: P4f[v][u][{i,f,g,o}] = (emb[v]@Wk_f + b_f) interleaved; Pb flat; maskv --------
__global__ __launch_bounds__(256) void precompute_kernel(
    const float* __restrict__ emb,
    const float* __restrict__ Wkf, const float* __restrict__ bf_,
    const float* __restrict__ Wkb, const float* __restrict__ bb_,
    float* __restrict__ P4f, float* __restrict__ Pb, int* __restrict__ maskv)
{
    int tid = threadIdx.x;
    for (int idx = tid; idx < Vn * G4; idx += 256) {
        int v = idx / G4, q = idx % G4;
        int u = q >> 2, k = q & 3;        // unit, gate
        int g = k * 64 + u;               // source gate column
        float sf = bf_[g];
        float sb = bb_[g];
        for (int d = 0; d < Dn; ++d) {
            float e = emb[v * Dn + d];
            sf += e * Wkf[d * G4 + g];
            sb += e * Wkb[d * G4 + g];
        }
        P4f[idx] = sf;                    // gate-interleaved for single b128 gather
        Pb[v * G4 + g] = sb;              // flat (only used in the head)
    }
    if (tid < Vn) {
        int m = 0;
        for (int d = 0; d < Dn; ++d) m |= (emb[tid * Dn + d] != 0.0f) ? 1 : 0;
        maskv[tid] = m;
    }
}

// -------- Main: forward LSTM recurrence on WMMA + fused backward step + head --------
__global__ __launch_bounds__(128) void bilstm_kernel(
    const int*   __restrict__ tokens,
    const float* __restrict__ Wrf,     // (64, 256) row-major
    const float* __restrict__ P4f_g,   // (6, 64, 4) interleaved
    const float* __restrict__ Pb_g,    // (6, 256)
    const int*   __restrict__ maskv_g, // (6,)
    const float* __restrict__ Wd,      // (128,)
    const float* __restrict__ bd,      // (1,)
    float* __restrict__ out)           // (256,)
{
    __shared__ __align__(16) __bf16 hbf[16][Un];   // h state in bf16 (A-matrix feed)
    __shared__ __align__(16) float  P4[Vn * G4];   // interleaved xproj table
    __shared__ int tokt[16];
    __shared__ int maskv[Vn];

    const int tid  = threadIdx.x;
    const int wave = tid >> 5;
    const int lane = tid & 31;
    const int hi   = (lane >> 4) & 1;
    const int lcol = lane & 15;
    const int u    = 16 * wave + lcol;       // this lane's hidden unit
    const int rowbase = blockIdx.x * 16;
    const int roff = hi ? 8 : 0;
    const int koff = hi ? 8 : 0;

    // ---- Load Wr_f once as bf16 B-fragments (32x16 per frag, 2 K-chunks x 4 gates) ----
    // B layout (ISA 7.12.2): elem e -> K = 32*kc + (hi?8:0) + e + (e&8), column fixed per lane.
    v16bf Bw[4][2];
    for (int g = 0; g < 4; ++g) {
        int col = g * 64 + u;
        for (int kc = 0; kc < 2; ++kc)
            for (int e = 0; e < 16; ++e) {
                int K = kc * 32 + (hi ? 8 : 0) + e + (e & 8);
                Bw[g][kc][e] = f2bf(Wrf[K * G4 + col]);
            }
    }

    for (int i = tid; i < Vn * G4; i += 128) P4[i] = P4f_g[i];
    if (tid < Vn) maskv[tid] = maskv_g[tid];
    for (int i = tid; i < (16 * Un * 2) / 4; i += 128) ((unsigned*)&hbf[0][0])[i] = 0u;

    v8f   cfrag = {};       // cell state (registers, C-matrix layout)
    float hfrag[8];         // lane-private copy of h (avoids LDS re-read for mask keep)
    for (int r = 0; r < 8; ++r) hfrag[r] = 0.0f;

    for (int t = 0; t < Tn; ++t) {
        if (tid < 16) tokt[tid] = tokens[(rowbase + tid) * Tn + t];
        __syncthreads();   // h(t-1), tokt visible

        // A fragments (16x32 bf16): lane's 16 elems are two contiguous 8-elem runs
        // (K = e + koff + (e&8)) -> two ds_load_b128 each, no conversions.
        const __bf16* hrow = &hbf[lcol][0];
        v8bf a0lo = *(const v8bf*)(hrow + koff);
        v8bf a0hi = *(const v8bf*)(hrow + koff + 16);
        v8bf a1lo = *(const v8bf*)(hrow + 32 + koff);
        v8bf a1hi = *(const v8bf*)(hrow + 32 + koff + 16);
        v16bf A0 = __builtin_shufflevector(a0lo, a0hi, 0,1,2,3,4,5,6,7,8,9,10,11,12,13,14,15);
        v16bf A1 = __builtin_shufflevector(a1lo, a1hi, 0,1,2,3,4,5,6,7,8,9,10,11,12,13,14,15);

        // Gather xproj biases straight into C-matrix layout: element r of gate g's
        // accumulator is exactly P[tok(row r)][g*64+u] -> feed as WMMA C operand.
        v8f cini[4];
        int toks[8];
        for (int r = 0; r < 8; ++r) {
            int tok = tokt[r + roff];
            toks[r] = tok;
            float4 p = *(const float4*)&P4[(tok * Un + u) * 4];
            cini[0][r] = p.x;
            cini[1][r] = p.y;
            cini[2][r] = p.z;
            cini[3][r] = p.w;
        }

        v8f acc[4];
        for (int g = 0; g < 4; ++g) {
            v8f z = __builtin_amdgcn_wmma_f32_16x16x32_bf16(false, A0, false, Bw[g][0],
                                                            (short)0, cini[g], false, false);
            acc[g] = __builtin_amdgcn_wmma_f32_16x16x32_bf16(false, A1, false, Bw[g][1],
                                                             (short)0, z, false, false);
        }

        // cell update: branch-free activations, mask-keep from registers
        for (int r = 0; r < 8; ++r) {
            float ig = fast_sigmoid(acc[0][r]);
            float fg = fast_sigmoid(acc[1][r]);
            float gg = fast_tanh   (acc[2][r]);
            float og = fast_sigmoid(acc[3][r]);
            float cn = fg * cfrag[r] + ig * gg;
            float hn = og * fast_tanh(cn);
            bool  mk = maskv[toks[r]] != 0;
            cfrag[r] = mk ? cn : cfrag[r];
            hfrag[r] = mk ? hn : hfrag[r];
        }
        __syncthreads();   // everyone done reading h(t-1)
        for (int r = 0; r < 8; ++r)
            hbf[r + roff][u] = f2bf(hfrag[r]);
    }
    __syncthreads();

    // ---- fused head: out[b] = h_f . Wd[0:64] + h_b . Wd[64:128] + bd ----
    // h_b is a single backward step from zero state on x[:,T-1] (h0@Wr_b == 0).
    if (tid < 16) {
        int row = tid;
        int b   = rowbase + row;
        int tok = tokens[b * Tn + (Tn - 1)];
        bool mk = maskv[tok] != 0;
        const float* P = &Pb_g[tok * G4];
        float accum = bd[0];
        for (int uu = 0; uu < Un; ++uu) {
            accum += bf2f(hbf[row][uu]) * Wd[uu];
            float cn = fast_sigmoid(P[uu]) * fast_tanh(P[128 + uu]);   // f*c0 = 0
            float hb = fast_sigmoid(P[192 + uu]) * fast_tanh(cn);
            accum += (mk ? hb : 0.0f) * Wd[64 + uu];
        }
        out[b] = accum;
    }
}

extern "C" void kernel_launch(void* const* d_in, const int* in_sizes, int n_in,
                              void* d_out, int out_size, void* d_ws, size_t ws_size,
                              hipStream_t stream) {
    (void)in_sizes; (void)n_in; (void)out_size; (void)ws_size;
    const int*   tokens = (const int*)  d_in[0];
    const float* emb    = (const float*)d_in[1];
    const float* Wkf    = (const float*)d_in[2];
    const float* Wrf    = (const float*)d_in[3];
    const float* bf_    = (const float*)d_in[4];
    const float* Wkb    = (const float*)d_in[5];
    const float* Wrb    = (const float*)d_in[6];
    const float* bb_    = (const float*)d_in[7];
    const float* Wd     = (const float*)d_in[8];
    const float* bd     = (const float*)d_in[9];
    (void)Wrb;  // backward direction only runs one step from h0=0 -> h0@Wr_b vanishes
    float* out = (float*)d_out;

    float* P4f   = (float*)d_ws;
    float* Pb    = P4f + Vn * G4;
    int*   maskv = (int*)(Pb + Vn * G4);

    precompute_kernel<<<1, 256, 0, stream>>>(emb, Wkf, bf_, Wkb, bb_, P4f, Pb, maskv);
    bilstm_kernel<<<Bsz / 16, 128, 0, stream>>>(tokens, Wrf, P4f, Pb, maskv, Wd, bd, out);
}
